// FlashMHA_25881472926068
// MI455X (gfx1250) — matrix operand order repaired and checked by reading, not compiled, not run on hardware
//
#include <hip/hip_runtime.h>

#define DI __device__ __forceinline__

typedef __attribute__((ext_vector_type(16))) __bf16 bf16x16;
typedef __attribute__((ext_vector_type(8)))  float  f32x8;

struct Bytes32 { uint4 a, b; };

static constexpr int Bn = 2, Tn = 2048, Sn = 2048, En = 1024, Hn = 16, Dn = 64;

DI unsigned short f2bf(float f) {
  unsigned u = __builtin_bit_cast(unsigned, f);
  u += 0x7FFFu + ((u >> 16) & 1u);          // round-to-nearest-even
  return (unsigned short)(u >> 16);
}

// Load a 32-byte WMMA fragment (16 bf16) as two aligned 16B segments.
DI bf16x16 ld_frag(const unsigned short* p0, const unsigned short* p1) {
  Bytes32 t;
  t.a = *(const uint4*)p0;
  t.b = *(const uint4*)p1;
  return __builtin_bit_cast(bf16x16, t);
}

DI f32x8 wmma_bf16(bf16x16 a, bf16x16 b, f32x8 c) {
  return __builtin_amdgcn_wmma_f32_16x16x32_bf16(
      /*neg_a=*/false, a, /*neg_b=*/false, b,
      /*c_mod=*/(short)0, c, /*reuse_a=*/false, /*reuse_b=*/false);
}

// ---------------------------------------------------------------------------
// Kernel 1: fused QKV projection.
//   X(4096x1024) @ W^T (W is [3072][1024] row-major) + bias
//   Output bf16, (B,H,T,D) layout per section (q/k/v selected by n-range).
// Block: 256 thr, tile 128(M) x 128(N), K-chunks of 32.
// ---------------------------------------------------------------------------
__global__ __launch_bounds__(256) void qkv_proj_kernel(
    const float* __restrict__ q, const float* __restrict__ k, const float* __restrict__ v,
    const float* __restrict__ W, const float* __restrict__ bias,
    unsigned short* __restrict__ qh, unsigned short* __restrict__ kh,
    unsigned short* __restrict__ vh)
{
  __shared__ unsigned short Asub[128 * 40];   // row stride 40 shorts = 80B (16B-aligned rows)
  __shared__ unsigned short Bsub[128 * 40];

  const int tid  = threadIdx.x;
  const int lane = tid & 31, wid = tid >> 5;
  const int wm = wid & 3, wn = wid >> 2;      // 4x2 wave grid: 32 M-rows x 64 N-cols per wave
  const int hf = lane >> 4, l16 = lane & 15;
  const int n_base = blockIdx.x * 128;
  const int m_base = blockIdx.y * 128;
  const int which  = n_base >> 10;            // 0=q, 1=k, 2=v (tiles never straddle 1024)
  const float* X = (which == 0) ? q : ((which == 1) ? k : v);
  unsigned short* Out = (which == 0) ? qh : ((which == 1) ? kh : vh);

  const f32x8 zero8 = {0.f,0.f,0.f,0.f,0.f,0.f,0.f,0.f};
  f32x8 acc[2][4];
#pragma unroll
  for (int fm = 0; fm < 2; ++fm)
#pragma unroll
    for (int fn = 0; fn < 4; ++fn) acc[fm][fn] = zero8;

  for (int kk = 0; kk < En; kk += 32) {
    __syncthreads();
#pragma unroll
    for (int i = 0; i < 4; ++i) {             // 1024 float4 per tile, 256 threads
      int f4  = tid + 256 * i;
      int row = f4 >> 3;
      int c4  = (f4 & 7) << 2;
      float4 av = *(const float4*)(X + (size_t)(m_base + row) * En + kk + c4);
      ushort4 as; as.x = f2bf(av.x); as.y = f2bf(av.y); as.z = f2bf(av.z); as.w = f2bf(av.w);
      *(ushort4*)(&Asub[row * 40 + c4]) = as;
      float4 bv = *(const float4*)(W + (size_t)(n_base + row) * En + kk + c4);
      ushort4 bs; bs.x = f2bf(bv.x); bs.y = f2bf(bv.y); bs.z = f2bf(bv.z); bs.w = f2bf(bv.w);
      *(ushort4*)(&Bsub[row * 40 + c4]) = bs;
    }
    __syncthreads();

    bf16x16 af[2], bfr[4];
#pragma unroll
    for (int fm = 0; fm < 2; ++fm) {
      int m = wm * 32 + fm * 16 + l16;
      af[fm] = ld_frag(&Asub[m * 40 + hf * 8], &Asub[m * 40 + 16 + hf * 8]);
    }
#pragma unroll
    for (int fn = 0; fn < 4; ++fn) {
      int n = wn * 64 + fn * 16 + l16;
      bfr[fn] = ld_frag(&Bsub[n * 40 + hf * 8], &Bsub[n * 40 + 16 + hf * 8]);
    }
#pragma unroll
    for (int fm = 0; fm < 2; ++fm)
#pragma unroll
      for (int fn = 0; fn < 4; ++fn)
        acc[fm][fn] = wmma_bf16(af[fm], bfr[fn], acc[fm][fn]);
  }

  // Epilogue: bias add + bf16 store in (B,H,T,D) layout.
#pragma unroll
  for (int fm = 0; fm < 2; ++fm) {
#pragma unroll
    for (int fn = 0; fn < 4; ++fn) {
      int n  = n_base + wn * 64 + fn * 16 + l16;
      int e  = n - (which << 10);
      int hh = e >> 6, d = e & 63;
      float bb = bias[n];
#pragma unroll
      for (int r = 0; r < 8; ++r) {
        int m = m_base + wm * 32 + fm * 16 + hf * 8 + r;
        int bidx = m >> 11;                   // / T
        int t    = m & (Tn - 1);
        Out[(((size_t)bidx * Hn + hh) * Tn + t) * Dn + d] = f2bf(acc[fm][fn][r] + bb);
      }
    }
  }
}

// ---------------------------------------------------------------------------
// Kernel 2: flash attention. grid = (T/128, B*H). Each wave: 16 Q-rows.
// Streams S in 32-wide tiles with online softmax; QK^T and PV via WMMA.
// ---------------------------------------------------------------------------
__global__ __launch_bounds__(256) void flash_attn_kernel(
    const unsigned short* __restrict__ qh, const unsigned short* __restrict__ kh,
    const unsigned short* __restrict__ vh, unsigned short* __restrict__ ctx)
{
  __shared__ unsigned short Klds[32 * 72];      // K tile row-major [s][d]  (== Btr for QK^T)
  __shared__ unsigned short Vt[64 * 40];        // V tile transposed [d][s] (== Btr for PV)
  __shared__ unsigned short Pl[8 * 16 * 40];    // per-wave P scratch (C-layout -> A-layout)

  const int tid  = threadIdx.x;
  const int lane = tid & 31, wid = tid >> 5;
  const int hf = lane >> 4, l16 = lane & 15;
  const int bh = blockIdx.y;
  const int b  = bh >> 4, h = bh & 15;
  const int t0 = blockIdx.x * 128;

  const unsigned short* Qb = qh + (size_t)bh * Tn * Dn;
  const unsigned short* Kb = kh + (size_t)bh * Sn * Dn;
  const unsigned short* Vb = vh + (size_t)bh * Sn * Dn;

  // Q fragments straight from global (row-major bf16 == A layout, 16B aligned).
  bf16x16 qa[2];
  {
    const unsigned short* qrow = Qb + (size_t)(t0 + wid * 16 + l16) * Dn;
#pragma unroll
    for (int kf = 0; kf < 2; ++kf) {
      Bytes32 tb;
      tb.a = *(const uint4*)(qrow + kf * 32 + hf * 8);
      tb.b = *(const uint4*)(qrow + kf * 32 + 16 + hf * 8);
      qa[kf] = __builtin_bit_cast(bf16x16, tb);
    }
  }

  const f32x8 zero8 = {0.f,0.f,0.f,0.f,0.f,0.f,0.f,0.f};
  f32x8 accD[4];
#pragma unroll
  for (int f = 0; f < 4; ++f) accD[f] = zero8;
  float rm[8], rl[8];
#pragma unroll
  for (int r = 0; r < 8; ++r) { rm[r] = -1e30f; rl[r] = 0.f; }

  unsigned short* Pw = &Pl[wid * 640];

  for (int sb = 0; sb < Sn; sb += 32) {
    __syncthreads();
    {
      int row = tid >> 3;
      int c8  = (tid & 7) << 3;
      uint4 kvq = *(const uint4*)(Kb + (size_t)(sb + row) * Dn + c8);
      *(uint4*)(&Klds[row * 72 + c8]) = kvq;
      union { uint4 u; unsigned short s[8]; } vv;
      vv.u = *(const uint4*)(Vb + (size_t)(sb + row) * Dn + c8);
#pragma unroll
      for (int j = 0; j < 8; ++j)
        Vt[(c8 + j) * 40 + row] = vv.s[j];    // transpose into [d][s]
    }
    __syncthreads();

    // scores = Q @ K^T  (16x32 per wave, two N-halves, K-dim = D = 64)
    f32x8 c0 = zero8, c1 = zero8;
#pragma unroll
    for (int kf = 0; kf < 2; ++kf) {
      int n0 = l16;
      bf16x16 kb0 = ld_frag(&Klds[n0 * 72 + kf * 32 + hf * 8],
                            &Klds[n0 * 72 + kf * 32 + 16 + hf * 8]);
      c0 = wmma_bf16(qa[kf], kb0, c0);
      int n1 = 16 + l16;
      bf16x16 kb1 = ld_frag(&Klds[n1 * 72 + kf * 32 + hf * 8],
                            &Klds[n1 * 72 + kf * 32 + 16 + hf * 8]);
      c1 = wmma_bf16(qa[kf], kb1, c1);
    }

    // Online softmax: row stats reduced across the 16-lane half-group.
#pragma unroll
    for (int r = 0; r < 8; ++r) {
      float sA = c0[r] * 0.125f;              // scale = D^-0.5
      float sB = c1[r] * 0.125f;
      float mx = fmaxf(sA, sB);
#pragma unroll
      for (int off = 1; off < 16; off <<= 1)
        mx = fmaxf(mx, __shfl_xor(mx, off, 32));
      float mn    = fmaxf(rm[r], mx);
      float alpha = __expf(rm[r] - mn);
      rm[r] = mn;
      float p0 = __expf(sA - mn);
      float p1 = __expf(sB - mn);
      float sum = p0 + p1;
#pragma unroll
      for (int off = 1; off < 16; off <<= 1)
        sum += __shfl_xor(sum, off, 32);
      rl[r] = rl[r] * alpha + sum;
      int mrow = hf * 8 + r;                  // C-layout row
      Pw[mrow * 40 + l16]      = f2bf(p0);
      Pw[mrow * 40 + 16 + l16] = f2bf(p1);
#pragma unroll
      for (int f = 0; f < 4; ++f)
        accD[f][r] *= alpha;
    }
    asm volatile("s_wait_dscnt 0x0" ::: "memory");   // P stores -> A-frag loads (same wave)

    bf16x16 pa = ld_frag(&Pw[l16 * 40 + hf * 8], &Pw[l16 * 40 + 16 + hf * 8]);
#pragma unroll
    for (int f = 0; f < 4; ++f) {
      int n = f * 16 + l16;
      bf16x16 vbf = ld_frag(&Vt[n * 40 + hf * 8], &Vt[n * 40 + 16 + hf * 8]);
      accD[f] = wmma_bf16(pa, vbf, accD[f]);
    }
  }

  // Normalize and write ctx as bf16 in (B,T,E) layout.
  int trow0 = t0 + wid * 16;
#pragma unroll
  for (int r = 0; r < 8; ++r) {
    int m = hf * 8 + r;
    float inv = 1.0f / rl[r];
    size_t base = ((size_t)b * Tn + (trow0 + m)) * En + h * 64;
#pragma unroll
    for (int f = 0; f < 4; ++f)
      ctx[base + f * 16 + l16] = f2bf(accD[f][r] * inv);
  }
}

// ---------------------------------------------------------------------------
// Kernel 3: output projection. ctx(bf16, 4096x1024) @ out_w^T + out_b -> f32.
// ---------------------------------------------------------------------------
__global__ __launch_bounds__(256) void out_proj_kernel(
    const unsigned short* __restrict__ ctx, const float* __restrict__ W,
    const float* __restrict__ bias, float* __restrict__ out)
{
  __shared__ unsigned short Asub[128 * 40];
  __shared__ unsigned short Bsub[128 * 40];

  const int tid  = threadIdx.x;
  const int lane = tid & 31, wid = tid >> 5;
  const int wm = wid & 3, wn = wid >> 2;
  const int hf = lane >> 4, l16 = lane & 15;
  const int n_base = blockIdx.x * 128;
  const int m_base = blockIdx.y * 128;

  const f32x8 zero8 = {0.f,0.f,0.f,0.f,0.f,0.f,0.f,0.f};
  f32x8 acc[2][4];
#pragma unroll
  for (int fm = 0; fm < 2; ++fm)
#pragma unroll
    for (int fn = 0; fn < 4; ++fn) acc[fm][fn] = zero8;

  for (int kk = 0; kk < En; kk += 32) {
    __syncthreads();
#pragma unroll
    for (int i = 0; i < 2; ++i) {             // A tile: 512 uint4 (bf16 source)
      int u   = tid + 256 * i;
      int row = u >> 2;
      int c8  = (u & 3) << 3;
      uint4 a = *(const uint4*)(ctx + (size_t)(m_base + row) * En + kk + c8);
      *(uint4*)(&Asub[row * 40 + c8]) = a;
    }
#pragma unroll
    for (int i = 0; i < 4; ++i) {             // B tile: 1024 float4 (f32 -> bf16)
      int f4  = tid + 256 * i;
      int row = f4 >> 3;
      int c4  = (f4 & 7) << 2;
      float4 bv = *(const float4*)(W + (size_t)(n_base + row) * En + kk + c4);
      ushort4 bs; bs.x = f2bf(bv.x); bs.y = f2bf(bv.y); bs.z = f2bf(bv.z); bs.w = f2bf(bv.w);
      *(ushort4*)(&Bsub[row * 40 + c4]) = bs;
    }
    __syncthreads();

    bf16x16 af[2], bfr[4];
#pragma unroll
    for (int fm = 0; fm < 2; ++fm) {
      int m = wm * 32 + fm * 16 + l16;
      af[fm] = ld_frag(&Asub[m * 40 + hf * 8], &Asub[m * 40 + 16 + hf * 8]);
    }
#pragma unroll
    for (int fn = 0; fn < 4; ++fn) {
      int n = wn * 64 + fn * 16 + l16;
      bfr[fn] = ld_frag(&Bsub[n * 40 + hf * 8], &Bsub[n * 40 + 16 + hf * 8]);
    }
#pragma unroll
    for (int fm = 0; fm < 2; ++fm)
#pragma unroll
      for (int fn = 0; fn < 4; ++fn)
        acc[fm][fn] = wmma_bf16(af[fm], bfr[fn], acc[fm][fn]);
  }

#pragma unroll
  for (int fm = 0; fm < 2; ++fm) {
#pragma unroll
    for (int fn = 0; fn < 4; ++fn) {
      int n = n_base + wn * 64 + fn * 16 + l16;
      float bb = bias[n];
#pragma unroll
      for (int r = 0; r < 8; ++r) {
        int m = m_base + wm * 32 + fm * 16 + hf * 8 + r;
        out[(size_t)m * En + n] = acc[fm][fn][r] + bb;
      }
    }
  }
}

// ---------------------------------------------------------------------------
extern "C" void kernel_launch(void* const* d_in, const int* in_sizes, int n_in,
                              void* d_out, int out_size, void* d_ws, size_t ws_size,
                              hipStream_t stream) {
  (void)in_sizes; (void)n_in; (void)out_size; (void)ws_size;
  const float* q     = (const float*)d_in[0];
  const float* k     = (const float*)d_in[1];
  const float* v     = (const float*)d_in[2];
  const float* w_in  = (const float*)d_in[3];
  const float* b_in  = (const float*)d_in[4];
  const float* w_out = (const float*)d_in[5];
  const float* b_out = (const float*)d_in[6];
  float* out = (float*)d_out;

  char* ws = (char*)d_ws;
  const size_t seg = (size_t)Bn * Tn * En * sizeof(unsigned short);  // 8 MB
  unsigned short* qh  = (unsigned short*)(ws);
  unsigned short* kh  = (unsigned short*)(ws + seg);
  unsigned short* vh  = (unsigned short*)(ws + 2 * seg);
  unsigned short* ctx = (unsigned short*)(ws + 3 * seg);

  qkv_proj_kernel<<<dim3(24, 32), 256, 0, stream>>>(q, k, v, w_in, b_in, qh, kh, vh);
  flash_attn_kernel<<<dim3(16, 32), 256, 0, stream>>>(qh, kh, vh, ctx);
  out_proj_kernel<<<dim3(8, 32), 256, 0, stream>>>(ctx, w_out, b_out, out);
}
